// WeightOnlyInt8Linear_21706764714504
// MI455X (gfx1250) — matrix operand order repaired and checked by reading, not compiled, not run on hardware
//
#include <hip/hip_runtime.h>

// ---- POD vector types (avoid HIP_vector_type union pitfalls) ----
typedef __attribute__((ext_vector_type(16))) _Float16 v16h;
typedef __attribute__((ext_vector_type(8)))  float    v8f;
typedef __attribute__((ext_vector_type(4)))  unsigned u32x4;
typedef __attribute__((ext_vector_type(4)))  float    f32x4;
typedef __attribute__((ext_vector_type(4)))  int      i32x4;

typedef __attribute__((address_space(1))) i32x4* gbl_i4p;  // global int4*
typedef __attribute__((address_space(3))) i32x4* lds_i4p;  // LDS int4*
typedef const __attribute__((address_space(1))) void* gbl_vp;

static constexpr int K    = 4096;   // IN_FEATURES
static constexpr int NOUT = 4096;   // OUT_FEATURES
static constexpr int MTOT = 4 * 2048;
static constexpr int GS   = 128;    // quant group size
static constexpr int BM = 128, BN = 128, BK = 64;
static constexpr int LDST = BK + 8; // LDS row stride in f16 (144 B, 16B-aligned, conflict-breaking)

union FragAB { u32x4 u[2]; v16h h; };
union H8     { _Float16 h[8]; u32x4 u; };

// ---- CDNA5 async global->LDS copy (16B per lane), ASYNCcnt-tracked ----
__device__ __forceinline__ void async_ld16(const _Float16* g, _Float16* l) {
#if __has_builtin(__builtin_amdgcn_global_load_async_to_lds_b128)
    // builtin signature (probe-discovered): (int4 AS1* src, int4 AS3* dst, imm offset, imm cpol)
    __builtin_amdgcn_global_load_async_to_lds_b128(
        (gbl_i4p)(size_t)(const void*)g,        // int round-trip sheds const; flat==global addr
        (lds_i4p)(i32x4*)l,                     // addrspacecast generic->LDS (addr[31:0])
        0, 0);
#else
    unsigned loff = (unsigned)(size_t)(const void*)l;  // addr[31:0] = LDS offset
    asm volatile("global_load_async_to_lds_b128 %0, %1, off"
                 :: "v"(loff), "v"((gbl_vp)(const void*)g) : "memory");
#endif
}

template <int N> __device__ __forceinline__ void wait_asynccnt() {
#if __has_builtin(__builtin_amdgcn_s_wait_asynccnt)
    __builtin_amdgcn_s_wait_asynccnt(N);
#else
    asm volatile("s_wait_asynccnt %0" :: "n"(N));
#endif
}

// ---- Prologue 1: input f32 -> f16 (8 elems / thread) ----
__global__ __launch_bounds__(256) void cvt_in_kernel(const float* __restrict__ x,
                                                     _Float16* __restrict__ y) {
    size_t f = ((size_t)blockIdx.x * blockDim.x + threadIdx.x) * 8;
    f32x4 a = *(const f32x4*)(x + f);
    f32x4 b = *(const f32x4*)(x + f + 4);
    H8 o;
    o.h[0] = (_Float16)a.x; o.h[1] = (_Float16)a.y;
    o.h[2] = (_Float16)a.z; o.h[3] = (_Float16)a.w;
    o.h[4] = (_Float16)b.x; o.h[5] = (_Float16)b.y;
    o.h[6] = (_Float16)b.z; o.h[7] = (_Float16)b.w;
    *(u32x4*)(y + f) = o.u;
}

// ---- Prologue 2: weight int32 * group scale -> f16 (8 elems / thread, one group) ----
__global__ __launch_bounds__(256) void dequant_w_kernel(const int* __restrict__ w,
                                                        const float* __restrict__ scales,
                                                        _Float16* __restrict__ wd) {
    size_t f = ((size_t)blockIdx.x * blockDim.x + threadIdx.x) * 8;
    int oc = (int)(f >> 12);          // / 4096
    int ii = (int)(f & (K - 1));      // % 4096
    float sc = scales[oc * (K / GS) + (ii / GS)];
    i32x4 a = *(const i32x4*)(w + f);
    i32x4 b = *(const i32x4*)(w + f + 4);
    H8 o;
    o.h[0] = (_Float16)((float)a.x * sc); o.h[1] = (_Float16)((float)a.y * sc);
    o.h[2] = (_Float16)((float)a.z * sc); o.h[3] = (_Float16)((float)a.w * sc);
    o.h[4] = (_Float16)((float)b.x * sc); o.h[5] = (_Float16)((float)b.y * sc);
    o.h[6] = (_Float16)((float)b.z * sc); o.h[7] = (_Float16)((float)b.w * sc);
    *(u32x4*)(wd + f) = o.u;
}

// ---- Main GEMM: C[m,n] = sum_k A[m,k] * Wd[n,k]; f16 WMMA, f32 accumulate ----
// Double-buffered LDS, async global->LDS staging overlapped with WMMA compute.
__global__ __launch_bounds__(256) void gemm_wmma_f16_kernel(const _Float16* __restrict__ A,
                                                            const _Float16* __restrict__ B,
                                                            float* __restrict__ C) {
    __shared__ _Float16 lsA[2][BM * LDST];
    __shared__ _Float16 lsB[2][BN * LDST];

    const int tid  = threadIdx.x;
    const int lane = tid & 31;
    const int l15  = lane & 15;
    const int hi   = lane >> 4;            // 0: lanes 0-15, 1: lanes 16-31
    const int wave = tid >> 5;             // 8 waves
    const int wM   = wave & 3;             // 4 waves along M
    const int wN   = wave >> 2;            // 2 waves along N
    const int bN0  = blockIdx.x * BN;
    const int bM0  = blockIdx.y * BM;
    const int m0   = wM * 32;              // wave tile: 32 (M) x 64 (N)
    const int n0   = wN * 64;

    // Per-thread staging chunks: 1024 16B chunks per tile, 4 per thread (A and B each)
    int srow[4], sch[4];
#pragma unroll
    for (int c = 0; c < 4; ++c) {
        int idx = tid + c * 256;
        srow[c] = idx >> 3;
        sch[c]  = (idx & 7) * 8;           // f16 offset of 16B chunk within row
    }

    v8f acc[2][4] = {};                    // eight 16x16 f32 accumulators

    // --- async issue of one BK tile (A+B) into buffer `buf` ---
    auto issue = [&](int buf, int kk) {
#pragma unroll
        for (int c = 0; c < 4; ++c) {
            async_ld16(A + (size_t)(bM0 + srow[c]) * K + kk + sch[c],
                       &lsA[buf][srow[c] * LDST + sch[c]]);
            async_ld16(B + (size_t)(bN0 + srow[c]) * K + kk + sch[c],
                       &lsB[buf][srow[c] * LDST + sch[c]]);
        }
    };

    constexpr int NIT = K / BK;            // 64
    issue(0, 0);

    for (int it = 0; it < NIT; ++it) {
        const int cur = it & 1;
        if (it + 1 < NIT) {
            issue(cur ^ 1, (it + 1) * BK); // prefetch next tile into other buffer
            wait_asynccnt<8>();            // in-order: previous tile's 8 ops are done
        } else {
            wait_asynccnt<0>();
        }
        __syncthreads();                   // all waves' tile data resident in LDS

        const _Float16* __restrict__ la = lsA[cur];
        const _Float16* __restrict__ lb = lsB[cur];
#pragma unroll
        for (int ks = 0; ks < 2; ++ks) {   // two K=32 sub-steps per staged BK=64
            FragAB fa[2];
            FragAB fb[4];
            // A fragment: lane l15 = row; 16 contiguous K at base = hi?16:0
            const int kbA = ks * 32 + hi * 16;
#pragma unroll
            for (int ti = 0; ti < 2; ++ti) {
                const _Float16* pa = &la[(m0 + ti * 16 + l15) * LDST + kbA];
                fa[ti].u[0] = *(const u32x4*)pa;
                fa[ti].u[1] = *(const u32x4*)(pa + 8);
            }
            // B fragment: lane l15 = column; 16 contiguous K at base = hi?16:0
            const int kbB = ks * 32 + hi * 16;
#pragma unroll
            for (int tj = 0; tj < 4; ++tj) {
                const _Float16* pb = &lb[(n0 + tj * 16 + l15) * LDST + kbB];
                fb[tj].u[0] = *(const u32x4*)pb;
                fb[tj].u[1] = *(const u32x4*)(pb + 8);
            }
#pragma unroll
            for (int ti = 0; ti < 2; ++ti)
#pragma unroll
                for (int tj = 0; tj < 4; ++tj)
                    acc[ti][tj] = __builtin_amdgcn_wmma_f32_16x16x32_f16(
                        false, fa[ti].h, false, fb[tj].h,
                        (short)0, acc[ti][tj], false, false);
        }
        __syncthreads();                   // all waves done reading `cur` buffer
    }

    // Epilogue: C/D layout — VGPR r, lanes 0-15: M=r, lanes 16-31: M=8+r; N = l15
#pragma unroll
    for (int ti = 0; ti < 2; ++ti)
#pragma unroll
        for (int tj = 0; tj < 4; ++tj)
#pragma unroll
            for (int r = 0; r < 8; ++r) {
                int row = bM0 + m0 + ti * 16 + hi * 8 + r;
                int col = bN0 + n0 + tj * 16 + l15;
                C[(size_t)row * NOUT + col] = acc[ti][tj][r];
            }
}

extern "C" void kernel_launch(void* const* d_in, const int* in_sizes, int n_in,
                              void* d_out, int out_size, void* d_ws, size_t ws_size,
                              hipStream_t stream) {
    const float* inp    = (const float*)d_in[0];   // (4,2048,4096) f32
    const int*   weight = (const int*)d_in[1];     // (4096,4096) int32 (int8-valued)
    const float* scales = (const float*)d_in[2];   // (4096,32) f32
    float*       out    = (float*)d_out;           // (4,2048,4096) f32

    _Float16* Ah = (_Float16*)d_ws;                                  // 67 MB
    _Float16* Wd = (_Float16*)((char*)d_ws + (size_t)MTOT * K * 2);  // +33.5 MB

    // input f32 -> f16 : 33,554,432 elems / 8 per thread
    cvt_in_kernel<<<(MTOT * (size_t)K) / (256 * 8), 256, 0, stream>>>(inp, Ah);
    // weight dequant -> f16 : 16,777,216 elems / 8 per thread
    dequant_w_kernel<<<((size_t)NOUT * K) / (256 * 8), 256, 0, stream>>>(weight, scales, Wd);
    // GEMM: grid = (N/BN, M/BM) = (32, 64)
    gemm_wmma_f16_kernel<<<dim3(NOUT / BN, MTOT / BM), 256, 0, stream>>>(Ah, Wd, out);
}
